// LSTMQNetwork_72404558676494
// MI455X (gfx1250) — matrix operand order repaired and checked by reading, not compiled, not run on hardware
//
#include <hip/hip_runtime.h>
#include <hip/hip_bf16.h>

// ---------------------------------------------------------------------------
// LSTM-Q network for MI455X (gfx1250, wave32, WMMA + TDM).
//
//   B=64, T=4096, D=128, H=512, G=4H=2048, A=18
//
// Kernel 0: transpose+convert Wi,Wh,Wv -> bf16, N-major (WMMA B-fragments are
//           two contiguous ds_load_b128 per lane).
// Kernel 1: x_proj = obses @ Wi + b (M=262144,N=2048,K=128), bf16 WMMA, fp32 out.
// Kernel 2: recurrent scan. 64 WGs = 4 batch slices (M=16) x 16 column groups.
//           Wh^T slice (128 KB) lives in LDS (loaded once by TDM); previous h
//           (16 KB contiguous) is TDM-loaded into LDS every step; 16 K-steps of
//           v_wmma_f32_16x16x32_bf16 per wave; sigmoid/tanh elementwise;
//           cross-WG sync via CDNA5 cluster barrier (s_barrier_signal/-wait -3).
// Kernel 3: qs = hs @ Wv + bv (N padded 18->32), bf16 WMMA.
// ---------------------------------------------------------------------------

#define DEVINL __device__ __forceinline__

typedef __attribute__((ext_vector_type(16))) __bf16       v16bf;
typedef __attribute__((ext_vector_type(8)))  float        v8f;
typedef __attribute__((ext_vector_type(4)))  unsigned int u32x4;
typedef __attribute__((ext_vector_type(2)))  unsigned int u32x2;
typedef __attribute__((ext_vector_type(4)))  float        f32x4;
typedef __attribute__((ext_vector_type(8)))  int          i32x8;
typedef __attribute__((ext_vector_type(4)))  int          i32x4;

static constexpr int CB = 64;      // batch
static constexpr int CT = 4096;    // time
static constexpr int CD = 128;     // obs dim
static constexpr int CH = 512;     // hidden
static constexpr int CG = 4 * CH;  // 2048 gate columns
static constexpr int CA = 18;      // actions
static constexpr int CM = CB * CT; // 262144 rows of x_proj / hs

union Frag16 { v16bf v; u32x4 q[2]; };

DEVINL unsigned short f2bf(float x) {
    unsigned int u = __float_as_uint(x);
    u += 0x7FFFu + ((u >> 16) & 1u);           // round-to-nearest-even
    return (unsigned short)(u >> 16);
}
DEVINL float fsigmoid(float x) { return 1.0f / (1.0f + __expf(-x)); }

// A-matrix fragment, 16x32 bf16, row-major in LDS (ld = K stride, elements).
// Layout: lane(half,m): e<8 -> K = 8*half+e ; e>=8 -> K = 16+8*half+(e-8).
DEVINL v16bf load_frag_a(const unsigned short* lds, int ld, int m0, int k0, int lane) {
    const int hh = lane >> 4, m = lane & 15;
    const unsigned short* base = lds + (size_t)(m0 + m) * ld + k0 + hh * 8;
    Frag16 f;
    f.q[0] = *(const u32x4*)(base);        // K = k0 + 8*half .. +7
    f.q[1] = *(const u32x4*)(base + 16);   // K = k0 + 16 + 8*half .. +7
    return f.v;
}

// B-matrix fragment, 32x16 bf16, stored N-major in LDS (row n holds K's).
// Layout: lane(half,n): e<8 -> K = 8*half+e ; e>=8 -> K = 16+8*half+(e-8).
DEVINL v16bf load_frag_b(const unsigned short* lds, int ld, int n0, int k0, int lane) {
    const int hh = lane >> 4, n = lane & 15;
    const unsigned short* base = lds + (size_t)(n0 + n) * ld + k0 + hh * 8;
    Frag16 f;
    f.q[0] = *(const u32x4*)(base);
    f.q[1] = *(const u32x4*)(base + 16);
    return f.v;
}

DEVINL v8f wmma_bf16(v16bf a, v16bf b, v8f c) {
    return __builtin_amdgcn_wmma_f32_16x16x32_bf16(false, a, false, b,
                                                   (short)0, c, false, false);
}

// ------------------------- TDM descriptor + issue --------------------------
// Builds a D# (ISA 8.3/8.4) for a 2D tile of 8-byte elements and issues
// TENSOR_LOAD_TO_LDS.  tile_d0/stride in 8-byte units; lds_off in bytes.
// This toolchain exposes the 6-arg builtin:
//   (uint32x4 g0, int32x8 g1, int32x4 g2, int32x4 g3, int32x8 g4, i32 cpol)
#if __has_builtin(__builtin_amdgcn_tensor_load_to_lds)
#define HAVE_TDM 1
DEVINL void tdm_load_to_lds(unsigned lds_off, const void* gptr,
                            unsigned tile_d0, unsigned tile_d1,
                            unsigned stride_d0) {
    const unsigned long long ga = (unsigned long long)(__SIZE_TYPE__)gptr;
    u32x4 g0;
    g0.x = 1u;                                   // count=1, user descriptor
    g0.y = lds_off;                              // lds_addr (bytes)
    g0.z = (unsigned)ga;                         // global_addr[31:0]
    g0.w = (unsigned)(ga >> 32) | (2u << 30);    // global_addr[56:32] | type=2
    const unsigned td0 = 1u << 20;               // tensor_dim0 (OOB bound)
    const unsigned td1 = tile_d1;                // tensor_dim1
    i32x8 g1;
    g1[0] = (int)(3u << 16);                     // wg_mask=0, data_size=3 (8B)
    g1[1] = (int)((td0 & 0xFFFFu) << 16);        // tensor_dim0[15:0]
    g1[2] = (int)((td0 >> 16) | ((td1 & 0xFFFFu) << 16));
    g1[3] = (int)((td1 >> 16) | (tile_d0 << 16));// tile_dim0
    g1[4] = (int)(tile_d1 & 0xFFFFu);            // tile_dim1 (tile_dim2=0)
    g1[5] = (int)stride_d0;                      // tensor_dim0_stride[31:0]
    g1[6] = 0;
    g1[7] = 0;
    const i32x4 z4 = {0, 0, 0, 0};
    const i32x8 z8 = {0, 0, 0, 0, 0, 0, 0, 0};
    __builtin_amdgcn_tensor_load_to_lds(g0, g1, z4, z4, z8, 0);
}
#else
#define HAVE_TDM 0
#endif

// ------------------------- kernel 0: weight prep ---------------------------
__global__ void prep_weights(const float* __restrict__ Wi,   // (D, G)
                             const float* __restrict__ Wh,   // (H, G)
                             const float* __restrict__ Wv,   // (H, A)
                             unsigned short* __restrict__ WiT,   // (G, D) bf16
                             unsigned short* __restrict__ WhT,   // (G, H) bf16
                             unsigned short* __restrict__ WvT) { // (32, H) bf16
    const size_t gt = (size_t)blockIdx.x * blockDim.x + threadIdx.x;
    if (gt < (size_t)CG * CH) {                  // WhT
        const int n = (int)(gt >> 9), k = (int)(gt & 511);
        WhT[gt] = f2bf(Wh[(size_t)k * CG + n]);
    }
    if (gt < (size_t)CG * CD) {                  // WiT
        const int n = (int)(gt >> 7), k = (int)(gt & 127);
        WiT[gt] = f2bf(Wi[(size_t)k * CG + n]);
    }
    if (gt < (size_t)32 * CH) {                  // WvT (pad 18 -> 32)
        const int n = (int)(gt >> 9), k = (int)(gt & 511);
        WvT[gt] = (n < CA) ? f2bf(Wv[(size_t)k * CA + n]) : (unsigned short)0;
    }
}

// --------------------- kernel 1: x_proj = obses@Wi + b ---------------------
__global__ void __launch_bounds__(256)
xproj_gemm(const float* __restrict__ obses,          // (B*T, D) fp32
           const unsigned short* __restrict__ WiT,   // (G, D) bf16
           const float* __restrict__ bias,           // (G)
           float* __restrict__ xproj) {              // (B*T, G) fp32
    __shared__ unsigned short aT[64 * CD];   // 16 KB
    __shared__ unsigned short bT[128 * CD];  // 32 KB
    const int tid  = threadIdx.x;
    const int lane = tid & 31, w = tid >> 5;
    const size_t m0 = (size_t)(blockIdx.x >> 4) * 64;
    const int    n0 = (int)(blockIdx.x & 15) * 128;

    // stage A: 64x128 fp32 -> bf16 LDS (row-major, K contiguous)
    #pragma unroll
    for (int i = 0; i < 8; ++i) {
        const int f4  = tid + i * 256;             // float4 index
        const int row = f4 >> 5, c4 = f4 & 31;
        f32x4 v = *(const f32x4*)(obses + (m0 + row) * CD + c4 * 4);
        u32x2 p;
        p.x = (unsigned)f2bf(v.x) | ((unsigned)f2bf(v.y) << 16);
        p.y = (unsigned)f2bf(v.z) | ((unsigned)f2bf(v.w) << 16);
        *(u32x2*)(aT + row * CD + c4 * 4) = p;
    }
    // stage B: contiguous 32 KB slab of WiT rows [n0, n0+128)
    #pragma unroll
    for (int i = 0; i < 8; ++i) {
        const int q = tid + i * 256;
        *(u32x4*)(bT + q * 8) = *(const u32x4*)(WiT + (size_t)n0 * CD + q * 8);
    }
    __syncthreads();

    const int hh = lane >> 4, nn = lane & 15;
    #pragma unroll
    for (int i = 0; i < 4; ++i) {
        const int tile = w * 4 + i;
        const int mt = tile >> 3, nt = tile & 7;
        const float bv0 = bias[n0 + nt * 16 + nn];
        v8f acc;
        #pragma unroll
        for (int r = 0; r < 8; ++r) acc[r] = bv0;
        #pragma unroll
        for (int ks = 0; ks < 4; ++ks) {
            v16bf af = load_frag_a(aT, CD, mt * 16, ks * 32, lane);
            v16bf bf = load_frag_b(bT, CD, nt * 16, ks * 32, lane);
            acc = wmma_bf16(af, bf, acc);
        }
        #pragma unroll
        for (int r = 0; r < 8; ++r) {
            const size_t row = m0 + mt * 16 + r + 8 * hh;
            xproj[row * CG + n0 + nt * 16 + nn] = acc[r];
        }
    }
}

// ---------------------- kernel 2: recurrent LSTM scan ----------------------
// 64 WGs: blockIdx = bs*16 + cg.  Cluster = the 16 consecutive cg-WGs of one
// batch slice.  Dynamic LDS = 157696 B.
__global__ void __launch_bounds__(256)
lstm_scan(const float* __restrict__ xproj,           // (B*T, G) fp32
          const unsigned short* __restrict__ WhT,    // (G, H) bf16
          const float* __restrict__ init_c,          // (B, H)
          const float* __restrict__ init_h,          // (B, H)
          unsigned short* __restrict__ hbuf,         // 2 * (B, H) bf16 ping-pong
          unsigned short* __restrict__ hs,           // (B, T, H) bf16 history
          float* __restrict__ out_c,                 // (B, H) fp32
          float* __restrict__ out_h) {               // (B, H) fp32
    extern __shared__ char smem[];
    unsigned short* whs   = (unsigned short*)(smem);            // 128x512 bf16 = 128 KB
    unsigned short* hprev = (unsigned short*)(smem + 131072);   // 16x512 bf16 = 16 KB
    float*          gates = (float*)(smem + 147456);            // 16x128 fp32 = 8 KB
    float*          cbuf  = (float*)(smem + 155648);            // 16x32  fp32 = 2 KB

    const int tid  = threadIdx.x;
    const int lane = tid & 31, w = tid >> 5;
    const int hh = lane >> 4, nn = lane & 15;
    const int bs = blockIdx.x >> 4;        // batch slice (cluster id)
    const int cg = blockIdx.x & 15;        // column group within cluster
    const int b0 = bs * 16;

    // ---- stage Wh^T slice into LDS (once): 4 contiguous 32 KB row-blocks,
    //      block g starts at global row g*512 + cg*32  ->  2D TDM tile.
#if HAVE_TDM
    if (w == 0) {
        // tile: 4 rows x 4096 (8B units); row stride = 512*512*2/8 = 65536.
        tdm_load_to_lds(0u, WhT + (size_t)cg * 32 * CH, 4096u, 4u, 65536u);
    }
#else
    #pragma unroll 4
    for (int i = 0; i < 32; ++i) {
        const int cidx = tid + i * 256;          // 16-byte chunk index
        const int lg = cidx >> 6, off = cidx & 63;
        const int grow = (lg >> 5) * CH + cg * 32 + (lg & 31);
        *(u32x4*)(whs + lg * CH + off * 8) =
            *(const u32x4*)(WhT + (size_t)grow * CH + off * 8);
    }
#endif
    // ---- init c (LDS) and h (ping buffer 0)
    #pragma unroll
    for (int k = 0; k < 2; ++k) {
        const int idx = tid + k * 256;           // 0..511
        const int row = idx >> 5, col = idx & 31;
        const size_t gidx = (size_t)(b0 + row) * CH + cg * 32 + col;
        cbuf[idx] = init_c[gidx];
        hbuf[gidx] = f2bf(init_h[gidx]);
    }
#if HAVE_TDM
    if (w == 0) __builtin_amdgcn_s_wait_tensorcnt(0);
#endif
    __threadfence();
    __syncthreads();
    if (w == 0) asm volatile("s_barrier_signal -3" ::: "memory");
    asm volatile("s_barrier_wait -3" ::: "memory");
    __threadfence();

    // this wave's gate tile: gate g = w>>1, local cols (w&1)*16 .. +15
    const int lcol = w * 16 + nn;                                 // 0..127 local
    const int gcol = (w >> 1) * CH + cg * 32 + (w & 1) * 16 + nn; // global col

    for (int t = 0; t < CT; ++t) {
        const unsigned short* cur = hbuf + (size_t)(t & 1) * CB * CH;
        unsigned short*       nxt = hbuf + (size_t)((t + 1) & 1) * CB * CH;

        // stage previous h: contiguous 16 KB (rows b0..b0+15, full K=512)
#if HAVE_TDM
        if (w == 0) {
            tdm_load_to_lds(131072u, cur + (size_t)b0 * CH, 2048u, 1u, 2048u);
            __builtin_amdgcn_s_wait_tensorcnt(0);
        }
#else
        #pragma unroll
        for (int i = 0; i < 4; ++i) {
            const int q = tid + i * 256;
            *(u32x4*)(hprev + q * 8) =
                *(const u32x4*)(cur + (size_t)b0 * CH + q * 8);
        }
#endif
        // accumulator initialized from x_proj (C-matrix layout)
        v8f acc;
        #pragma unroll
        for (int r = 0; r < 8; ++r) {
            const size_t xrow = (size_t)(b0 + r + 8 * hh) * CT + t;
            acc[r] = xproj[xrow * CG + gcol];
        }
        if (t + 1 < CT) {  // prefetch next step's x_proj column
            const size_t xrow = (size_t)(b0 + 8 * hh) * CT + (t + 1);
            __builtin_prefetch(&xproj[xrow * CG + gcol], 0, 1);
        }
        __syncthreads();   // h_prev tile visible to all waves

        // gates_tile += h_prev @ Wh_slice   (K = 512 -> 16 WMMA k-steps)
        #pragma unroll
        for (int ks = 0; ks < 16; ++ks) {
            v16bf af = load_frag_a(hprev, CH, 0, ks * 32, lane);
            v16bf bf = load_frag_b(whs, CH, w * 16, ks * 32, lane);
            acc = wmma_bf16(af, bf, acc);
        }
        #pragma unroll
        for (int r = 0; r < 8; ++r)
            gates[(r + 8 * hh) * 128 + lcol] = acc[r];
        __syncthreads();

        // elementwise gate math on 16x32 c/h slice (2 elements / thread)
        #pragma unroll
        for (int k = 0; k < 2; ++k) {
            const int idx = tid + k * 256;
            const int row = idx >> 5, col = idx & 31;
            const float iv = fsigmoid(gates[row * 128 + col]);
            const float fv = fsigmoid(gates[row * 128 + 32 + col]);
            const float gv = tanhf(gates[row * 128 + 64 + col]);
            const float ov = fsigmoid(gates[row * 128 + 96 + col]);
            const float cn = fv * cbuf[idx] + iv * gv;
            const float hn = ov * tanhf(cn);
            cbuf[idx] = cn;
            const unsigned short hb = f2bf(hn);
            const size_t gidx = (size_t)(b0 + row) * CH + cg * 32 + col;
            nxt[gidx] = hb;
            hs[((size_t)(b0 + row) * CT + t) * CH + cg * 32 + col] = hb;
            if (t == CT - 1) { out_c[gidx] = cn; out_h[gidx] = hn; }
        }
        __threadfence();           // release this step's h to the cluster (L2)
        __syncthreads();
        if (w == 0) asm volatile("s_barrier_signal -3" ::: "memory");
        asm volatile("s_barrier_wait -3" ::: "memory");
        __threadfence();           // acquire other WGs' h writes
    }
}

// ----------------------- kernel 3: qs = hs @ Wv + bv -----------------------
__global__ void __launch_bounds__(256)
qhead_gemm(const unsigned short* __restrict__ hs,    // (B*T, H) bf16
           const unsigned short* __restrict__ WvT,   // (32, H) bf16
           const float* __restrict__ bv,             // (A)
           float* __restrict__ qs) {                 // (B*T, A) fp32
    __shared__ unsigned short aT[64 * 128];  // 16 KB (one K-chunk)
    __shared__ unsigned short bT[32 * CH];   // 32 KB (all of Wv^T)
    const int tid  = threadIdx.x;
    const int lane = tid & 31, w = tid >> 5;
    const int hh = lane >> 4, nn = lane & 15;
    const size_t m0 = (size_t)blockIdx.x * 64;
    const int mt = w >> 1, nt = w & 1;
    const int ncol = nt * 16 + nn;

    #pragma unroll
    for (int i = 0; i < 8; ++i) {    // stage Wv^T (contiguous 32 KB)
        const int q = tid + i * 256;
        *(u32x4*)(bT + q * 8) = *(const u32x4*)(WvT + (size_t)q * 8);
    }

    v8f acc;
    const float b0 = (ncol < CA) ? bv[ncol] : 0.0f;
    #pragma unroll
    for (int r = 0; r < 8; ++r) acc[r] = b0;

    for (int kc = 0; kc < 4; ++kc) {         // K = 512 in 4 chunks of 128
        __syncthreads();
        #pragma unroll
        for (int i = 0; i < 4; ++i) {        // stage 64x128 bf16 A chunk
            const int q = tid + i * 256;     // 16-byte chunks
            const int row = q >> 4, off = q & 15;
            *(u32x4*)(aT + row * 128 + off * 8) =
                *(const u32x4*)(hs + (m0 + row) * CH + kc * 128 + off * 8);
        }
        __syncthreads();
        #pragma unroll
        for (int ks = 0; ks < 4; ++ks) {
            v16bf af = load_frag_a(aT, 128, mt * 16, ks * 32, lane);
            v16bf bf = load_frag_b(bT, CH, nt * 16, kc * 128 + ks * 32, lane);
            acc = wmma_bf16(af, bf, acc);
        }
    }
    if (ncol < CA) {
        #pragma unroll
        for (int r = 0; r < 8; ++r) {
            const size_t row = m0 + mt * 16 + r + 8 * hh;
            qs[row * CA + ncol] = acc[r];
        }
    }
}

// ------------------------------- launcher ----------------------------------
extern "C" void kernel_launch(void* const* d_in, const int* in_sizes, int n_in,
                              void* d_out, int out_size, void* d_ws, size_t ws_size,
                              hipStream_t stream) {
    const float* obses  = (const float*)d_in[0];
    const float* init_c = (const float*)d_in[1];
    const float* init_h = (const float*)d_in[2];
    const float* Wi     = (const float*)d_in[3];
    const float* Wh     = (const float*)d_in[4];
    const float* b      = (const float*)d_in[5];
    const float* Wv     = (const float*)d_in[6];
    const float* bv     = (const float*)d_in[7];

    char* ws = (char*)d_ws;
    // workspace layout (bytes), 256-aligned
    unsigned short* WiT  = (unsigned short*)(ws + 0);           //   512 KB
    unsigned short* WhT  = (unsigned short*)(ws + 524288);      //     2 MB
    unsigned short* WvT  = (unsigned short*)(ws + 2621440);     //    32 KB
    unsigned short* hbuf = (unsigned short*)(ws + 2654208);     //   128 KB (ping-pong h)
    unsigned short* hsb  = (unsigned short*)(ws + 2785280);     //   256 MB (h history bf16)
    float*          xprj = (float*)(ws + 271220736);            //  2048 MB (x_proj fp32)

    float* out_c = (float*)d_out;                 // (B,H)
    float* out_h = (float*)d_out + 32768;         // (B,H)
    float* qs    = (float*)d_out + 65536;         // (B,T,A)

    prep_weights<<<4096, 256, 0, stream>>>(Wi, Wh, Wv, WiT, WhT, WvT);
    xproj_gemm<<<(CM / 64) * (CG / 128), 256, 0, stream>>>(obses, WiT, b, xprj);
    // 64 WGs; dispatch should form 16-WG clusters over consecutive blockIdx.
    lstm_scan<<<64, 256, 157696, stream>>>(xprj, WhT, init_c, init_h,
                                           hbuf, hsb, out_c, out_h);
    qhead_gemm<<<CM / 64, 256, 0, stream>>>(hsb, WvT, bv, qs);
}
